// NER_62912680952629
// MI455X (gfx1250) — compile-verified, matrix operand order checked by reading, not run on hardware
//
#include <hip/hip_runtime.h>
#include <hip/hip_bf16.h>
#include <stdint.h>

// ---------------------------------------------------------------------------
// MI455X (gfx1250) fused NER head — v2 with 2x register blocking.
//   B=8, S=2048, H=768, TAG=9, E=128, HEADS=12, dh=64  -> scale 1/96
// All GEMMs on v_wmma_f32_16x16x32_bf16 (wave32). Fused GLU avoids 300MB of
// u/v intermediates. 2 M-tiles per wave (GLU) and 2x2 tiles (proj/score)
// halve L2 weight traffic and improve load:wmma ratio.
// ---------------------------------------------------------------------------

typedef __attribute__((ext_vector_type(16))) __bf16 v16bf;
typedef __attribute__((ext_vector_type(8)))  __bf16 v8bf;
typedef __attribute__((ext_vector_type(8)))  float  v8f;

#define BQ 8
#define SQ 2048
#define HQ 768
#define H3 2304
#define TAGQ 9
#define EQ 128
#define NEGV (-50000.0f)

// ---- WMMA fragment loader (ISA 7.12.2 layout, wave32) ---------------------
// A-fragment 16x32 bf16 from row-major:
//   lane L (<16):  row=L, K={k0..k0+7, k0+16..k0+23}
//   lane L+16:     row=L, K={k0+8..k0+15, k0+24..k0+31}
// Same loader on W^T (row-major [N][K]) yields the B-fragment.
static __device__ __forceinline__ v16bf load_frag(const __hip_bfloat16* base,
                                                  int ld, int r0, int k0,
                                                  int lane) {
  int row = r0 + (lane & 15);
  int kofs = k0 + ((lane >> 4) << 3);          // +0 or +8
  const __hip_bfloat16* p = base + (size_t)row * ld + kofs;
  v8bf lo = *(const v8bf*)(p);
  v8bf hi = *(const v8bf*)(p + 16);
  return __builtin_shufflevector(lo, hi, 0, 1, 2, 3, 4, 5, 6, 7,
                                 8, 9, 10, 11, 12, 13, 14, 15);
}

static __device__ __forceinline__ v8f wmma_bf16(v16bf a, v16bf b, v8f c) {
  return __builtin_amdgcn_wmma_f32_16x16x32_bf16(false, a, false, b,
                                                 (short)0, c, false, false);
}

// ---- pack / transpose / gather kernels ------------------------------------
__global__ void cvt_bf16_kernel(const float* __restrict__ src,
                                __hip_bfloat16* __restrict__ dst, int n) {
  int i = blockIdx.x * blockDim.x + threadIdx.x;
  if (i < n) dst[i] = __float2bfloat16(src[i]);
}

// src [R][C] f32 row-major -> dst [C][R] bf16 row-major (src^T)
__global__ void transpose_bf16_kernel(const float* __restrict__ src,
                                      __hip_bfloat16* __restrict__ dst,
                                      int R, int C) {
  int i = blockIdx.x * blockDim.x + threadIdx.x;
  if (i < R * C) {
    int r = i / C, c = i % C;
    dst[(size_t)c * R + r] = __float2bfloat16(src[i]);
  }
}

// Wo [2304][9] -> WoT [16][2304] bf16, tags 9..15 zero-padded
__global__ void wo_pack_kernel(const float* __restrict__ Wo,
                               __hip_bfloat16* __restrict__ WoT) {
  int i = blockIdx.x * blockDim.x + threadIdx.x;
  if (i < 16 * H3) {
    int t = i / H3, c = i % H3;
    float v = (t < TAGQ) ? Wo[(size_t)c * TAGQ + t] : 0.0f;
    WoT[i] = __float2bfloat16(v);
  }
}

// entB[b,e,:] = bf16(hidden[b, entity_start[b,e], :])
__global__ void gather_entity_kernel(const float* __restrict__ hid,
                                     const int* __restrict__ ent,
                                     __hip_bfloat16* __restrict__ entB) {
  int i = blockIdx.x * blockDim.x + threadIdx.x;
  if (i < BQ * EQ * HQ) {
    int h = i % HQ;
    int be = i / HQ;
    int b = be / EQ;
    int s = ent[be];
    entB[i] = __float2bfloat16(hid[((size_t)b * SQ + s) * HQ + h]);
  }
}

// ---- fused GLU + output projection ----------------------------------------
// start_logit = (sigmoid(hs@Wu+bu) * (hs@Wv+bv)) @ Wo + bo
// Each wave owns 32 rows (2 M-tiles). Per 32-column chunk j of the 2304 dim:
// 8 WMMA accumulators (u/v x 2 cols x 2 Mtiles) over K=768, gate in-register,
// bounce both 16x32 bf16 gate tiles through LDS to A-layout, 2 WMMAs vs WoT.
__global__ __launch_bounds__(128) void glu_start_kernel(
    const __hip_bfloat16* __restrict__ hsB, const __hip_bfloat16* __restrict__ WuT,
    const __hip_bfloat16* __restrict__ WvT, const __hip_bfloat16* __restrict__ WoT,
    const float* __restrict__ bu, const float* __restrict__ bv,
    const float* __restrict__ bo, float* __restrict__ start_out) {
  __shared__ __hip_bfloat16 gtile[4][2][16][40];  // 40: rows stay 16B aligned
  int lane = threadIdx.x & 31;
  int wave = threadIdx.x >> 5;
  int r0 = (blockIdx.x * 4 + wave) * 32;          // 2 M-tiles: r0, r0+16
  int col = lane & 15;
  int rbase = (lane >> 4) << 3;                   // +0 or +8 (C/D layout)

  v8f accA = {}, accB = {};                       // start_logit accumulators
  for (int j = 0; j < H3; j += 32) {
    v8f u0a = {}, u1a = {}, v0a = {}, v1a = {};
    v8f u0b = {}, u1b = {}, v0b = {}, v1b = {};
    for (int k = 0; k < HQ; k += 32) {
      v16bf aA = load_frag(hsB, HQ, r0, k, lane);
      v16bf aB = load_frag(hsB, HQ, r0 + 16, k, lane);
      __builtin_prefetch(hsB + (size_t)r0 * HQ + k + 64, 0, 3);
      v16bf b0 = load_frag(WuT, HQ, j, k, lane);
      v16bf b1 = load_frag(WuT, HQ, j + 16, k, lane);
      v16bf b2 = load_frag(WvT, HQ, j, k, lane);
      v16bf b3 = load_frag(WvT, HQ, j + 16, k, lane);
      u0a = wmma_bf16(aA, b0, u0a);  u0b = wmma_bf16(aB, b0, u0b);
      u1a = wmma_bf16(aA, b1, u1a);  u1b = wmma_bf16(aB, b1, u1b);
      v0a = wmma_bf16(aA, b2, v0a);  v0b = wmma_bf16(aB, b2, v0b);
      v1a = wmma_bf16(aA, b3, v1a);  v1b = wmma_bf16(aB, b3, v1b);
    }
    float bu0 = bu[j + col], bu1 = bu[j + 16 + col];
    float bv0 = bv[j + col], bv1 = bv[j + 16 + col];
#pragma unroll
    for (int i = 0; i < 8; ++i) {
      int row = rbase + i;
      float uu, vv;
      uu = u0a[i] + bu0; vv = v0a[i] + bv0;
      gtile[wave][0][row][col]      = __float2bfloat16(vv / (1.0f + __expf(-uu)));
      uu = u1a[i] + bu1; vv = v1a[i] + bv1;
      gtile[wave][0][row][16 + col] = __float2bfloat16(vv / (1.0f + __expf(-uu)));
      uu = u0b[i] + bu0; vv = v0b[i] + bv0;
      gtile[wave][1][row][col]      = __float2bfloat16(vv / (1.0f + __expf(-uu)));
      uu = u1b[i] + bu1; vv = v1b[i] + bv1;
      gtile[wave][1][row][16 + col] = __float2bfloat16(vv / (1.0f + __expf(-uu)));
    }
    __syncthreads();                              // uniform over 4 waves
    v16bf wb = load_frag(WoT, H3, 0, j, lane);    // 16 tag cols, K-chunk j
    v16bf gA = load_frag(&gtile[wave][0][0][0], 40, 0, 0, lane);
    v16bf gB = load_frag(&gtile[wave][1][0][0], 40, 0, 0, lane);
    accA = wmma_bf16(gA, wb, accA);
    accB = wmma_bf16(gB, wb, accB);
    __syncthreads();
  }
  if (col < TAGQ) {
    float bb = bo[col];
#pragma unroll
    for (int i = 0; i < 8; ++i) {
      start_out[(size_t)(r0 + rbase + i) * TAGQ + col]      = accA[i] + bb;
      start_out[(size_t)(r0 + 16 + rbase + i) * TAGQ + col] = accB[i] + bb;
    }
  }
}

// ---- projection: Y_bf16[M][768] = bf16(X_bf16 @ W + b), W given as W^T ----
// 2x2 tile blocking per wave: 32 rows x 32 cols.
__global__ __launch_bounds__(128) void proj_bf16_kernel(
    const __hip_bfloat16* __restrict__ X, const __hip_bfloat16* __restrict__ WT,
    const float* __restrict__ bias, __hip_bfloat16* __restrict__ Y) {
  int lane = threadIdx.x & 31;
  int wave = threadIdx.x >> 5;
  int n0 = (blockIdx.x * 4 + wave) * 32;          // 2 column tiles
  int r0 = blockIdx.y * 32;                       // 2 row tiles
  v8f c00 = {}, c01 = {}, c10 = {}, c11 = {};
  for (int k = 0; k < HQ; k += 32) {
    v16bf aA = load_frag(X, HQ, r0, k, lane);
    v16bf aB = load_frag(X, HQ, r0 + 16, k, lane);
    v16bf bA = load_frag(WT, HQ, n0, k, lane);
    v16bf bB = load_frag(WT, HQ, n0 + 16, k, lane);
    c00 = wmma_bf16(aA, bA, c00);
    c01 = wmma_bf16(aA, bB, c01);
    c10 = wmma_bf16(aB, bA, c10);
    c11 = wmma_bf16(aB, bB, c11);
  }
  int col = lane & 15;
  int rbase = (lane >> 4) << 3;
  float b0 = bias[n0 + col], b1 = bias[n0 + 16 + col];
#pragma unroll
  for (int i = 0; i < 8; ++i) {
    Y[(size_t)(r0 + rbase + i) * HQ + n0 + col]           = __float2bfloat16(c00[i] + b0);
    Y[(size_t)(r0 + rbase + i) * HQ + n0 + 16 + col]      = __float2bfloat16(c01[i] + b1);
    Y[(size_t)(r0 + 16 + rbase + i) * HQ + n0 + col]      = __float2bfloat16(c10[i] + b0);
    Y[(size_t)(r0 + 16 + rbase + i) * HQ + n0 + 16 + col] = __float2bfloat16(c11[i] + b1);
  }
}

// ---- attention scores: end[b,e,s] = (q[b,e,:]·k[b,s,:])/96, masked --------
// 2x2 tile blocking per wave: 32 e-rows x 32 s-cols.
__global__ __launch_bounds__(128) void score_kernel(
    const __hip_bfloat16* __restrict__ qB, const __hip_bfloat16* __restrict__ kB,
    const uint8_t* __restrict__ mask, float* __restrict__ end_out) {
  int lane = threadIdx.x & 31;
  int wave = threadIdx.x >> 5;
  int s0 = (blockIdx.x * 4 + wave) * 32;          // 2 s tiles
  int e0 = blockIdx.y * 32;                       // 2 e tiles
  int b = blockIdx.z;
  const __hip_bfloat16* q = qB + (size_t)b * EQ * HQ;
  const __hip_bfloat16* kk = kB + (size_t)b * SQ * HQ;
  v8f c00 = {}, c01 = {}, c10 = {}, c11 = {};
  for (int k = 0; k < HQ; k += 32) {
    v16bf aA = load_frag(q, HQ, e0, k, lane);
    v16bf aB = load_frag(q, HQ, e0 + 16, k, lane);
    v16bf bA = load_frag(kk, HQ, s0, k, lane);
    v16bf bB = load_frag(kk, HQ, s0 + 16, k, lane);
    __builtin_prefetch(kk + (size_t)s0 * HQ + k + 64, 0, 3);
    c00 = wmma_bf16(aA, bA, c00);
    c01 = wmma_bf16(aA, bB, c01);
    c10 = wmma_bf16(aB, bA, c10);
    c11 = wmma_bf16(aB, bB, c11);
  }
  int col = lane & 15;
  int rbase = (lane >> 4) << 3;
  int sA = s0 + col, sB = s0 + 16 + col;
  bool okA = mask[(size_t)b * SQ + sA] != 0;
  bool okB = mask[(size_t)b * SQ + sB] != 0;
  const float inv = 1.0f / 96.0f;                 // 1/(HEADS*sqrt(dh))
#pragma unroll
  for (int i = 0; i < 8; ++i) {
    size_t rowA = (size_t)b * EQ + e0 + rbase + i;
    size_t rowB = rowA + 16;
    end_out[rowA * SQ + sA] = okA ? c00[i] * inv : NEGV;
    end_out[rowA * SQ + sB] = okB ? c01[i] * inv : NEGV;
    end_out[rowB * SQ + sA] = okA ? c10[i] * inv : NEGV;
    end_out[rowB * SQ + sB] = okB ? c11[i] * inv : NEGV;
  }
}

// ---------------------------------------------------------------------------
extern "C" void kernel_launch(void* const* d_in, const int* in_sizes, int n_in,
                              void* d_out, int out_size, void* d_ws, size_t ws_size,
                              hipStream_t stream) {
  const float* hid = (const float*)d_in[0];
  const float* Wu  = (const float*)d_in[1];
  const float* bu  = (const float*)d_in[2];
  const float* Wv  = (const float*)d_in[3];
  const float* bv  = (const float*)d_in[4];
  const float* Wo  = (const float*)d_in[5];
  const float* bo  = (const float*)d_in[6];
  const float* Wq  = (const float*)d_in[7];
  const float* bq  = (const float*)d_in[8];
  const float* Wk  = (const float*)d_in[9];
  const float* bk  = (const float*)d_in[10];
  const uint8_t* mask = (const uint8_t*)d_in[11];
  const int* ent   = (const int*)d_in[12];

  float* start_out = (float*)d_out;                       // [8,2048,9]
  float* end_out   = start_out + (size_t)BQ * SQ * TAGQ;  // [8,128,2048]

  // workspace layout (bf16), all 16B aligned
  char* w = (char*)d_ws;
  __hip_bfloat16* hsB  = (__hip_bfloat16*)(w);                       // 16384x768
  __hip_bfloat16* entB = (__hip_bfloat16*)(w + 25165824);            // 1024x768
  __hip_bfloat16* WuT  = (__hip_bfloat16*)(w + 26738688);            // 2304x768
  __hip_bfloat16* WvT  = (__hip_bfloat16*)(w + 30277632);            // 2304x768
  __hip_bfloat16* WkT  = (__hip_bfloat16*)(w + 33816576);            // 768x768
  __hip_bfloat16* WqT  = (__hip_bfloat16*)(w + 34996224);            // 768x768
  __hip_bfloat16* WoT  = (__hip_bfloat16*)(w + 36175872);            // 16x2304
  __hip_bfloat16* kB   = (__hip_bfloat16*)(w + 36249600);            // 16384x768
  __hip_bfloat16* qB   = (__hip_bfloat16*)(w + 61415424);            // 1024x768

  const int T = 256;
  int nHs = BQ * SQ * HQ;
  cvt_bf16_kernel<<<(nHs + T - 1) / T, T, 0, stream>>>(hid, hsB, nHs);
  transpose_bf16_kernel<<<(HQ * H3 + T - 1) / T, T, 0, stream>>>(Wu, WuT, HQ, H3);
  transpose_bf16_kernel<<<(HQ * H3 + T - 1) / T, T, 0, stream>>>(Wv, WvT, HQ, H3);
  transpose_bf16_kernel<<<(HQ * HQ + T - 1) / T, T, 0, stream>>>(Wk, WkT, HQ, HQ);
  transpose_bf16_kernel<<<(HQ * HQ + T - 1) / T, T, 0, stream>>>(Wq, WqT, HQ, HQ);
  wo_pack_kernel<<<(16 * H3 + T - 1) / T, T, 0, stream>>>(Wo, WoT);
  gather_entity_kernel<<<(BQ * EQ * HQ + T - 1) / T, T, 0, stream>>>(hid, ent, entB);

  // fused GLU + Wo : 16384 rows, 128 rows per block (4 waves x 32)
  glu_start_kernel<<<dim3((BQ * SQ) / 128), dim3(128), 0, stream>>>(
      hsB, WuT, WvT, WoT, bu, bv, bo, start_out);

  // k = hs@Wk+bk (16384 rows), q = ent@Wq+bq (1024 rows), bf16 outputs
  proj_bf16_kernel<<<dim3(HQ / 128, (BQ * SQ) / 32), dim3(128), 0, stream>>>(
      hsB, WkT, bk, kB);
  proj_bf16_kernel<<<dim3(HQ / 128, (BQ * EQ) / 32), dim3(128), 0, stream>>>(
      entB, WqT, bq, qB);

  // end logits: per-batch 128x2048 = q.k^T / 96 with mask
  score_kernel<<<dim3(SQ / 128, EQ / 32, BQ), dim3(128), 0, stream>>>(
      qB, kB, mask, end_out);
}
